// CausalSelfAttention_16853451669899
// MI455X (gfx1250) — matrix operand order repaired and checked by reading, not compiled, not run on hardware
//
#include <hip/hip_runtime.h>
#include <hip/hip_bf16.h>

// ---------------- problem constants ----------------
#define EMB    1024
#define LSEQ   2048
#define NHEAD  16
#define HDIM   64
#define NBATCH 2
#define NQKV   3072                 // 3*EMB
#define MROWS  (NBATCH * LSEQ)      // 4096

// ---------------- GEMM tiling ----------------
#define BM  128
#define BN  128
#define BK  32
#define BKP 40                      // padded LDS k-stride (elems) -> 80B rows

typedef __attribute__((ext_vector_type(16))) __bf16 v16bf;
typedef __attribute__((ext_vector_type(8)))  float  v8f;

union FragU {
  v16bf v;
  uint4 q[2];
};

// Native bf16 convert (gfx1250 has bf16 datapath; fptrunc is RTNE).
__device__ __forceinline__ unsigned short f2bf(float x) {
  __bf16 h = (__bf16)x;
  return __builtin_bit_cast(unsigned short, h);
}

// ---------------- gfx1250 async global->LDS copy (ASYNCcnt-tracked) --------
#if defined(__gfx1250__) && __has_builtin(__builtin_amdgcn_global_load_async_to_lds_b128)
#define HAVE_ASYNC_LDS 1
#else
#define HAVE_ASYNC_LDS 0
#endif

#if HAVE_ASYNC_LDS
typedef int vint4 __attribute__((vector_size(16)));
__device__ __forceinline__ void async_cp16(const void* g, void* l) {
  __builtin_amdgcn_global_load_async_to_lds_b128(
      (__attribute__((address_space(1))) vint4*)g,
      (__attribute__((address_space(3))) vint4*)l, 0, 0);
}
__device__ __forceinline__ void wait_async0() {
#if __has_builtin(__builtin_amdgcn_s_wait_asynccnt)
  __builtin_amdgcn_s_wait_asynccnt(0);
#else
  asm volatile("s_wait_asynccnt 0x0" ::: "memory");
#endif
}
#else
__device__ __forceinline__ void wait_async0() {}
#endif

__device__ __forceinline__ void cp16(const void* g, void* l) {
  *reinterpret_cast<uint4*>(l) = *reinterpret_cast<const uint4*>(g);
}

// A-matrix (16xK) bf16 fragment. Lane holds row M = lane&15; `slab` points at
// that row's 32 contiguous K elements.  Per ISA: VGPR j<4 -> K = hi*8 + 2j,
// VGPR j>=4 -> K = 16 + hi*8 + 2(j-4).  => two 16B loads at hi*16 and 32+hi*16.
__device__ __forceinline__ v16bf load_frag_a(const unsigned short* slab, int hi) {
  FragU f;
  const uint4* p = reinterpret_cast<const uint4*>(slab);
  f.q[0] = p[hi];
  f.q[1] = p[2 + hi];
  return f.v;
}

// B-matrix (Kx16) bf16 fragment. Lane holds column N = lane&15; `slab` points
// at that column's 32 contiguous K elements.  Same K order as the A fragment:
// two 16B loads at hi*16 and 32+hi*16.
__device__ __forceinline__ v16bf load_frag_b(const unsigned short* slab, int hi) {
  FragU f;
  const uint4* p = reinterpret_cast<const uint4*>(slab);
  f.q[0] = p[hi];
  f.q[1] = p[2 + hi];
  return f.v;
}

__device__ __forceinline__ v8f wmma_bf16(v16bf a, v16bf b, v8f c) {
  // (neg_a, A, neg_b, B, c_mod, C, reuse_a, reuse_b)
  return __builtin_amdgcn_wmma_f32_16x16x32_bf16(false, a, false, b, (short)0, c,
                                                 false, false);
}

// =====================================================================
// Kernel 1: QKV projection  C[4096,3072] = X[4096,1024] @ Wqkv + b
// Scatters results as bf16 into per-head Q[B,H,L,D], K[B,H,L,D], Vt[B,H,D,L].
// =====================================================================
__global__ __launch_bounds__(256) void qkv_gemm_kernel(
    const float* __restrict__ X, const float* __restrict__ W,
    const float* __restrict__ bias,
    unsigned short* __restrict__ Qbf, unsigned short* __restrict__ Kbf,
    unsigned short* __restrict__ Vt) {
  __shared__ __align__(16) unsigned short As[BM * BKP];
  __shared__ __align__(16) unsigned short Bs[BN * BKP];
  const int tid  = threadIdx.x;
  const int wave = tid >> 5;
  const int lane = tid & 31;
  const int ln = lane & 15, hi = lane >> 4;
  const int m0 = blockIdx.y * BM;
  const int n0 = blockIdx.x * BN;
  const int wm = (wave & 1) * 64;      // wave 64x32 sub-tile
  const int wn = (wave >> 1) * 32;

  v8f acc[4][2];
#pragma unroll
  for (int mt = 0; mt < 4; ++mt)
#pragma unroll
    for (int nt = 0; nt < 2; ++nt) acc[mt][nt] = {};

  for (int k0 = 0; k0 < EMB; k0 += BK) {
    // stage A 128x32 (f32 -> bf16), coalesced float4 loads
#pragma unroll
    for (int i = 0; i < 4; ++i) {
      int f = tid + i * 256;                 // 1024 float4
      int r = f >> 3, c = (f & 7) * 4;
      const float* src = X + (size_t)(m0 + r) * EMB + k0 + c;
      if (k0 + BK < EMB) __builtin_prefetch(src + BK, 0, 3);
      float4 v = *reinterpret_cast<const float4*>(src);
      unsigned short* d = &As[r * BKP + c];
      d[0] = f2bf(v.x); d[1] = f2bf(v.y); d[2] = f2bf(v.z); d[3] = f2bf(v.w);
    }
    // stage B 32x128 transposed into [n][k] (f32 -> bf16)
#pragma unroll
    for (int i = 0; i < 4; ++i) {
      int f = tid + i * 256;
      int k = f >> 5, c = (f & 31) * 4;
      const float* src = W + (size_t)(k0 + k) * NQKV + n0 + c;
      if (k0 + BK < EMB) __builtin_prefetch(src + (size_t)BK * NQKV, 0, 3);
      float4 v = *reinterpret_cast<const float4*>(src);
      Bs[(c + 0) * BKP + k] = f2bf(v.x);
      Bs[(c + 1) * BKP + k] = f2bf(v.y);
      Bs[(c + 2) * BKP + k] = f2bf(v.z);
      Bs[(c + 3) * BKP + k] = f2bf(v.w);
    }
    __syncthreads();
    v16bf bfr[2];
#pragma unroll
    for (int nt = 0; nt < 2; ++nt)
      bfr[nt] = load_frag_b(&Bs[(wn + nt * 16 + ln) * BKP], hi);
#pragma unroll
    for (int mt = 0; mt < 4; ++mt) {
      v16bf afr = load_frag_a(&As[(wm + mt * 16 + ln) * BKP], hi);
#pragma unroll
      for (int nt = 0; nt < 2; ++nt)
        acc[mt][nt] = wmma_bf16(afr, bfr[nt], acc[mt][nt]);
    }
    __syncthreads();
  }

  // epilogue: bias + scatter into Q / K / V^T (reference packs per-head 3D chunks)
#pragma unroll
  for (int mt = 0; mt < 4; ++mt)
#pragma unroll
    for (int nt = 0; nt < 2; ++nt)
#pragma unroll
      for (int v = 0; v < 8; ++v) {
        int gm = m0 + wm + mt * 16 + v + 8 * hi;   // C layout: M = v + 8*hi
        int gc = n0 + wn + nt * 16 + ln;           // N = lane&15
        float val = acc[mt][nt][v] + bias[gc];
        int b = gm >> 11;                          // / LSEQ
        int l = gm & (LSEQ - 1);
        int head = gc / (3 * HDIM);
        int wi = gc % (3 * HDIM);
        unsigned short bv = f2bf(val);
        size_t bh = (size_t)b * NHEAD + head;
        if (wi < HDIM)
          Qbf[(bh * LSEQ + l) * HDIM + wi] = bv;
        else if (wi < 2 * HDIM)
          Kbf[(bh * LSEQ + l) * HDIM + (wi - HDIM)] = bv;
        else
          Vt[(bh * HDIM + (wi - 2 * HDIM)) * LSEQ + l] = bv;
      }
}

// =====================================================================
// Kernel 2: fused flash attention per (b, h, 128 q-rows).
// 8 waves x 16 q-rows; 32-key blocks staged once per workgroup in LDS via
// async global->LDS copies; online softmax; P re-layouted through LDS.
// Reference mask semantics: causal AND rows q >= mask_len[b] fully masked
// (=> uniform softmax = mean(V)); both fall out of the same -1e18 logit path.
// =====================================================================
#define PSTR 40   // LDS P-tile k-stride (elems)

__global__ __launch_bounds__(256) void attn_kernel(
    const unsigned short* __restrict__ Qbf, const unsigned short* __restrict__ Kbf,
    const unsigned short* __restrict__ Vt, const int* __restrict__ mask_len,
    unsigned short* __restrict__ Ybf) {
  __shared__ __align__(16) unsigned short Ks[32 * HDIM];       // [key][d]  4KB
  __shared__ __align__(16) unsigned short Vs[HDIM * 32];       // [d][key]  4KB
  __shared__ __align__(16) unsigned short Plds[8 * 16 * PSTR]; // 10KB
  const int tid = threadIdx.x, wave = tid >> 5, lane = tid & 31;
  const int ln = lane & 15, hi = lane >> 4;
  const int b = blockIdx.y >> 4;
  const int h = blockIdx.y & 15;
  const int q0 = blockIdx.x * 128 + wave * 16;
  const int mlen = mask_len[b];

  const unsigned short* Qh = Qbf + ((size_t)b * NHEAD + h) * LSEQ * HDIM;
  const unsigned short* Kh = Kbf + ((size_t)b * NHEAD + h) * LSEQ * HDIM;
  const unsigned short* Vh = Vt + ((size_t)b * NHEAD + h) * HDIM * LSEQ;
  unsigned short* Pw = &Plds[wave * 16 * PSTR];

  // Q fragments, resident for the whole kernel (row = q0 + ln, D = 2 k-steps)
  v16bf qf[2];
  qf[0] = load_frag_a(Qh + (size_t)(q0 + ln) * HDIM, hi);
  qf[1] = load_frag_a(Qh + (size_t)(q0 + ln) * HDIM + 32, hi);

  v8f o[4];
#pragma unroll
  for (int nt = 0; nt < 4; ++nt) o[nt] = {};
  float mrow[8], lrow[8];
#pragma unroll
  for (int v = 0; v < 8; ++v) { mrow[v] = -3.0e38f; lrow[v] = 0.0f; }

  // per-thread staging chunk coordinates (16B chunks, 256 of each)
  const int kr = tid >> 3, kd8 = (tid & 7) * 8;   // K: [32 keys][64 d]
  const int vd = tid >> 2, vk8 = (tid & 3) * 8;   // V: [64 d][32 keys]

  for (int kb = 0; kb < LSEQ; kb += 32) {
    // ---- stage K/V block for the whole workgroup (async -> LDS) ----
    {
      const unsigned short* ksrc = Kh + (size_t)(kb + kr) * HDIM + kd8;
      const unsigned short* vsrc = Vh + (size_t)vd * LSEQ + kb + vk8;
#if HAVE_ASYNC_LDS
      async_cp16(ksrc, &Ks[kr * HDIM + kd8]);
      async_cp16(vsrc, &Vs[vd * 32 + vk8]);
#else
      cp16(ksrc, &Ks[kr * HDIM + kd8]);
      cp16(vsrc, &Vs[vd * 32 + vk8]);
#endif
    }
    wait_async0();
    __syncthreads();

    // ---- S = Q @ K^T for 32 keys (two 16-key tiles) ----
    v8f s0 = {}, s1 = {};
#pragma unroll
    for (int ks = 0; ks < 2; ++ks) {
      v16bf k0f = load_frag_b(&Ks[(ln)*HDIM + ks * 32], hi);
      v16bf k1f = load_frag_b(&Ks[(16 + ln) * HDIM + ks * 32], hi);
      s0 = wmma_bf16(qf[ks], k0f, s0);
      s1 = wmma_bf16(qf[ks], k1f, s1);
    }
    // ---- mask + scale + online softmax ----
    const int key0 = kb + ln, key1 = kb + 16 + ln;
    float pv0[8], pv1[8], newmax[8];
#pragma unroll
    for (int v = 0; v < 8; ++v) {
      int qrow = q0 + v + 8 * hi;
      bool rvalid = qrow < mlen;
      float a0 = (rvalid && key0 <= qrow) ? s0[v] * 0.125f : -1.0e18f;
      float a1 = (rvalid && key1 <= qrow) ? s1[v] * 0.125f : -1.0e18f;
      pv0[v] = a0; pv1[v] = a1;
      float t = fmaxf(a0, a1);
#pragma unroll
      for (int off = 1; off < 16; off <<= 1) t = fmaxf(t, __shfl_xor(t, off, 32));
      newmax[v] = fmaxf(mrow[v], t);
    }
#pragma unroll
    for (int v = 0; v < 8; ++v) {
      float alpha = __expf(mrow[v] - newmax[v]);
      float p0 = __expf(pv0[v] - newmax[v]);
      float p1 = __expf(pv1[v] - newmax[v]);
      float rs = p0 + p1;
#pragma unroll
      for (int off = 1; off < 16; off <<= 1) rs += __shfl_xor(rs, off, 32);
      lrow[v] = lrow[v] * alpha + rs;
      mrow[v] = newmax[v];
#pragma unroll
      for (int nt = 0; nt < 4; ++nt) o[nt][v] *= alpha;
      int M = v + 8 * hi;                 // C-layout row -> LDS P[m][k]
      Pw[M * PSTR + ln] = f2bf(p0);
      Pw[M * PSTR + 16 + ln] = f2bf(p1);
    }
    // ---- O += P @ V  (P via LDS layout transpose; V^T gives contiguous K) ----
    v16bf pf = load_frag_a(Pw + ln * PSTR, hi);
#pragma unroll
    for (int nt = 0; nt < 4; ++nt) {
      v16bf vf = load_frag_b(&Vs[(nt * 16 + ln) * 32], hi);
      o[nt] = wmma_bf16(pf, vf, o[nt]);
    }
    __syncthreads();   // protect Ks/Vs before next block overwrites them
  }
  // ---- epilogue: O / l  -> Y[b, q, h*64 + d] (bf16 feed for out-proj) ----
#pragma unroll
  for (int v = 0; v < 8; ++v) {
    float inv = 1.0f / lrow[v];
    int qrow = q0 + v + 8 * hi;
#pragma unroll
    for (int nt = 0; nt < 4; ++nt) {
      float y = o[nt][v] * inv;
      Ybf[((size_t)b * LSEQ + qrow) * EMB + h * HDIM + nt * 16 + ln] = f2bf(y);
    }
  }
}

// =====================================================================
// Kernel 3: output projection  Out[4096,1024] = Y @ W_o + b_o  (f32 out)
// =====================================================================
__global__ __launch_bounds__(256) void out_gemm_kernel(
    const unsigned short* __restrict__ Ybf, const float* __restrict__ W,
    const float* __restrict__ bias, float* __restrict__ Out) {
  __shared__ __align__(16) unsigned short As[BM * BKP];
  __shared__ __align__(16) unsigned short Bs[BN * BKP];
  const int tid  = threadIdx.x;
  const int wave = tid >> 5;
  const int lane = tid & 31;
  const int ln = lane & 15, hi = lane >> 4;
  const int m0 = blockIdx.y * BM;
  const int n0 = blockIdx.x * BN;
  const int wm = (wave & 1) * 64;
  const int wn = (wave >> 1) * 32;

  v8f acc[4][2];
#pragma unroll
  for (int mt = 0; mt < 4; ++mt)
#pragma unroll
    for (int nt = 0; nt < 2; ++nt) acc[mt][nt] = {};

  for (int k0 = 0; k0 < EMB; k0 += BK) {
    // stage A 128x32 (already bf16): 512 x 16B copies, async -> LDS when able
#pragma unroll
    for (int i = 0; i < 2; ++i) {
      int f = tid + i * 256;
      int r = f >> 2, c = (f & 3) * 8;
      const unsigned short* src = Ybf + (size_t)(m0 + r) * EMB + k0 + c;
#if HAVE_ASYNC_LDS
      async_cp16(src, &As[r * BKP + c]);
#else
      cp16(src, &As[r * BKP + c]);
#endif
    }
    // stage B 32x128 transposed (f32 -> bf16)
#pragma unroll
    for (int i = 0; i < 4; ++i) {
      int f = tid + i * 256;
      int k = f >> 5, c = (f & 31) * 4;
      const float* src = W + (size_t)(k0 + k) * EMB + n0 + c;
      if (k0 + BK < EMB) __builtin_prefetch(src + (size_t)BK * EMB, 0, 3);
      float4 v = *reinterpret_cast<const float4*>(src);
      Bs[(c + 0) * BKP + k] = f2bf(v.x);
      Bs[(c + 1) * BKP + k] = f2bf(v.y);
      Bs[(c + 2) * BKP + k] = f2bf(v.z);
      Bs[(c + 3) * BKP + k] = f2bf(v.w);
    }
    wait_async0();
    __syncthreads();
    v16bf bfr[2];
#pragma unroll
    for (int nt = 0; nt < 2; ++nt)
      bfr[nt] = load_frag_b(&Bs[(wn + nt * 16 + ln) * BKP], hi);
#pragma unroll
    for (int mt = 0; mt < 4; ++mt) {
      v16bf afr = load_frag_a(&As[(wm + mt * 16 + ln) * BKP], hi);
#pragma unroll
      for (int nt = 0; nt < 2; ++nt)
        acc[mt][nt] = wmma_bf16(afr, bfr[nt], acc[mt][nt]);
    }
    __syncthreads();
  }
#pragma unroll
  for (int mt = 0; mt < 4; ++mt)
#pragma unroll
    for (int nt = 0; nt < 2; ++nt)
#pragma unroll
      for (int v = 0; v < 8; ++v) {
        int gm = m0 + wm + mt * 16 + v + 8 * hi;
        int gc = n0 + wn + nt * 16 + ln;
        Out[(size_t)gm * EMB + gc] = acc[mt][nt][v] + bias[gc];
      }
}

// =====================================================================
// launch
// =====================================================================
extern "C" void kernel_launch(void* const* d_in, const int* in_sizes, int n_in,
                              void* d_out, int out_size, void* d_ws, size_t ws_size,
                              hipStream_t stream) {
  const float* x      = (const float*)d_in[0];
  const float* W_qkv  = (const float*)d_in[1];
  const float* b_qkv  = (const float*)d_in[2];
  const float* W_o    = (const float*)d_in[3];
  const float* b_o    = (const float*)d_in[4];
  const int* mask_len = (const int*)d_in[5];

  // workspace layout (bf16): Q, K: [B,H,L,D]; Vt: [B,H,D,L]; Y: [B,L,E]
  // 4 x 8.39 MB = 33.6 MB total
  char* ws = (char*)d_ws;
  const size_t seg = (size_t)NBATCH * NHEAD * LSEQ * HDIM * sizeof(unsigned short);
  unsigned short* Qbf = (unsigned short*)(ws);
  unsigned short* Kbf = (unsigned short*)(ws + seg);
  unsigned short* Vt  = (unsigned short*)(ws + 2 * seg);
  unsigned short* Ybf = (unsigned short*)(ws + 3 * seg);

  dim3 blk(256);
  qkv_gemm_kernel<<<dim3(NQKV / BN, MROWS / BM), blk, 0, stream>>>(
      x, W_qkv, b_qkv, Qbf, Kbf, Vt);
  attn_kernel<<<dim3(LSEQ / 128, NBATCH * NHEAD), blk, 0, stream>>>(
      Qbf, Kbf, Vt, mask_len, Ybf);
  out_gemm_kernel<<<dim3(EMB / BN, MROWS / BM), blk, 0, stream>>>(
      Ybf, W_o, b_o, (float*)d_out);
}
